// mMHSA_20066087207506
// MI455X (gfx1250) — matrix operand order repaired and checked by reading, not compiled, not run on hardware
//
#include <hip/hip_runtime.h>

// mixed MHSA for MI455X (gfx1250, wave32).
// Stage 1 (x6 launches): projection GEMM f32->f16, WMMA f32_16x16x32_f16.
// Stage 2: dual-stream flash attention with double-buffered async global->LDS
//          K/V staging (GLOBAL_LOAD_ASYNC_TO_LDS_B128 + s_wait_asynccnt).
// Workspace: 6 f16 tensors of 4096x1024 = 48 MB.

typedef _Float16 v16h __attribute__((ext_vector_type(16)));
typedef _Float16 v8h  __attribute__((ext_vector_type(8)));
typedef float    v8f  __attribute__((ext_vector_type(8)));
typedef int      v4i  __attribute__((ext_vector_type(4)));

// address-space-qualified element types (so pointer casts keep the AS)
typedef v4i __attribute__((address_space(1))) v4i_g;   // global
typedef v4i __attribute__((address_space(3))) v4i_l;   // LDS

union AF { v16h v; v8h h[2]; };

#if __has_builtin(__builtin_amdgcn_global_load_async_to_lds_b128) && \
    __has_builtin(__builtin_amdgcn_s_wait_asynccnt)
#define USE_ASYNC 1
#else
#define USE_ASYNC 0
#endif

__device__ __forceinline__ void copy16_to_lds(_Float16* dst, const _Float16* src) {
#if USE_ASYNC
    __builtin_amdgcn_global_load_async_to_lds_b128((v4i_g*)src, (v4i_l*)dst, 0, 0);
#else
    *reinterpret_cast<uint4*>(dst) = *reinterpret_cast<const uint4*>(src);
#endif
}

__device__ __forceinline__ void wait_tile_done(bool more_inflight) {
#if USE_ASYNC
    if (more_inflight) __builtin_amdgcn_s_wait_asynccnt(4);  // next tile's 4 ops remain
    else               __builtin_amdgcn_s_wait_asynccnt(0);
#else
    (void)more_inflight;
#endif
}

__device__ __forceinline__ v8f vzero8() {
    v8f z;
#pragma unroll
    for (int i = 0; i < 8; ++i) z[i] = 0.0f;
    return z;
}

__device__ __forceinline__ v8f wmma16(const AF& a, const AF& b, v8f c) {
    return __builtin_amdgcn_wmma_f32_16x16x32_f16(
        /*neg_a=*/false, a.v, /*neg_b=*/false, b.v,
        /*c_mod=*/(short)0, c, /*reuse_a=*/false, /*reuse_b=*/false);
}

// ---------------------------------------------------------------------------
// Stage 1: Y = x @ W^T. x:[4096,1024] f32, W:[1024,1024] f32. One launch/proj.
// Output (f16): Q/K type -> [b*16+h][n][64];  V type -> [b*16+h][64][n].
// ---------------------------------------------------------------------------
#define P_BM 128
#define P_BK 32
#define P_LD 40   // padded LDS row stride (halves), 80B rows (16B-aligned)

__global__ __launch_bounds__(256) void proj_kernel(
    const float* __restrict__ x, const float* __restrict__ W,
    _Float16* __restrict__ out, const int isV)
{
    __shared__ _Float16 Xs[P_BM][P_LD];
    __shared__ _Float16 Wt[P_BM][P_LD];

    const int tid  = threadIdx.x;
    const int lane = tid & 31;
    const int wave = tid >> 5;
    const int lo   = lane & 15;
    const int hi   = lane >> 4;          // 0 or 1
    const int m0   = blockIdx.y * P_BM;  // 0..4095
    const int n0   = blockIdx.x * P_BM;  // 0..1023

    v8f acc[8];
#pragma unroll
    for (int i = 0; i < 8; ++i) acc[i] = vzero8();

    for (int k0 = 0; k0 < 1024; k0 += P_BK) {
        __syncthreads();
        // cooperative load + f32->f16 convert: 128x32 for X and W
#pragma unroll
        for (int i = 0; i < 4; ++i) {
            int e   = tid + 256 * i;     // 0..1023
            int row = e >> 3;            // 0..127
            int c4  = e & 7;             // 8 float4 per row
            float4 xv = reinterpret_cast<const float4*>(x + (size_t)(m0 + row) * 1024 + k0)[c4];
            _Float16* dx = &Xs[row][c4 * 4];
            dx[0] = (_Float16)xv.x; dx[1] = (_Float16)xv.y;
            dx[2] = (_Float16)xv.z; dx[3] = (_Float16)xv.w;
            float4 wv = reinterpret_cast<const float4*>(W + (size_t)(n0 + row) * 1024 + k0)[c4];
            _Float16* dw = &Wt[row][c4 * 4];
            dw[0] = (_Float16)wv.x; dw[1] = (_Float16)wv.y;
            dw[2] = (_Float16)wv.z; dw[3] = (_Float16)wv.w;
            if (k0 + P_BK < 1024) {     // global_prefetch_b8 for next K chunk
                __builtin_prefetch(x + (size_t)(m0 + row) * 1024 + k0 + P_BK, 0, 1);
                __builtin_prefetch(W + (size_t)(n0 + row) * 1024 + k0 + P_BK, 0, 1);
            }
        }
        __syncthreads();

        // A fragment: rows wave*16 .. +15, K = 16*(e>>3) + 8*hi + (e&7)
        AF a;
        a.h[0] = *reinterpret_cast<const v8h*>(&Xs[wave * 16 + lo][hi * 8]);
        a.h[1] = *reinterpret_cast<const v8h*>(&Xs[wave * 16 + lo][16 + hi * 8]);
#pragma unroll
        for (int nt = 0; nt < 8; ++nt) {
            // B fragment: col = nt*16+lo, K = 16*(e>>3) + 8*hi + (e&7)
            AF b;
            b.h[0] = *reinterpret_cast<const v8h*>(&Wt[nt * 16 + lo][hi * 8]);
            b.h[1] = *reinterpret_cast<const v8h*>(&Wt[nt * 16 + lo][16 + hi * 8]);
            acc[nt] = wmma16(a, b, acc[nt]);
        }
    }

    // Scatter-store f16 results in attention-friendly layout.
#pragma unroll
    for (int nt = 0; nt < 8; ++nt) {
        int nn = n0 + nt * 16 + lo;       // 0..1023
        int hh = nn >> 6, dh = nn & 63;
#pragma unroll
        for (int r = 0; r < 8; ++r) {
            int m  = m0 + wave * 16 + r + hi * 8;   // C layout: row = r + 8*hi
            int bb = m >> 11, nq = m & 2047;
            size_t off = isV ? ((size_t)(bb * 16 + hh) * 64 + dh) * 2048 + nq
                             : ((size_t)(bb * 16 + hh) * 2048 + nq) * 64 + dh;
            out[off] = (_Float16)acc[nt][r];
        }
    }
}

// ---------------------------------------------------------------------------
// Stage 2: dual flash attention. One WG per (b*16+h, q-block of 128).
// 8 waves x 16 q-rows each; K/V tiled by 32 along sequence, double-buffered
// via async global->LDS loads.
// ---------------------------------------------------------------------------
#define A_KT 32
#define A_NT 64   // 2048 / A_KT tiles

__global__ __launch_bounds__(256) void attn_kernel(
    const _Float16* __restrict__ ws, float* __restrict__ outp)
{
    __shared__ _Float16 Ks[2][2][A_KT][64];    // [buf][stream][j][d]
    __shared__ _Float16 Vts[2][2][64][A_KT];   // [buf][stream][dh][j]
    __shared__ _Float16 Pbuf[8][16][A_KT];     // per-wave softmax tile scratch

    const int tid  = threadIdx.x;
    const int lane = tid & 31;
    const int wave = tid >> 5;
    const int lo   = lane & 15;
    const int hi   = lane >> 4;
    const int bh   = blockIdx.y;           // b*16 + h
    const int b    = bh >> 4;
    const int h    = bh & 15;
    const int q0   = blockIdx.x * 128;
    const int a1   = h & 1;                // score stream feeding v1 (swap on odd heads)

    const size_t SZ = 4096ull * 1024ull;
    const size_t HO = (size_t)bh * 2048 * 64;   // per-(b,h) offset (same for both layouts)
    const _Float16* __restrict__ Qp = ws + 0 * SZ + HO;
    const _Float16* __restrict__ Kp = ws + 1 * SZ + HO;
    const _Float16* __restrict__ Vp = ws + 2 * SZ + HO;
    const _Float16* __restrict__ Qc = ws + 3 * SZ + HO;
    const _Float16* __restrict__ Kc = ws + 4 * SZ + HO;
    const _Float16* __restrict__ Vc = ws + 5 * SZ + HO;

    // Q A-fragments, loaded once directly from global (16B chunks).
    AF qa[2][2];
    {
        const _Float16* Qs[2] = {Qp, Qc};
        const int row = q0 + wave * 16 + lo;
#pragma unroll
        for (int s = 0; s < 2; ++s)
#pragma unroll
            for (int dc = 0; dc < 2; ++dc) {
                const _Float16* base = Qs[s] + (size_t)row * 64 + dc * 32;
                qa[s][dc].h[0] = *reinterpret_cast<const v8h*>(base + hi * 8);
                qa[s][dc].h[1] = *reinterpret_cast<const v8h*>(base + 16 + hi * 8);
            }
    }

    v8f O1[4], O2[4];
#pragma unroll
    for (int i = 0; i < 4; ++i) { O1[i] = vzero8(); O2[i] = vzero8(); }
    float mst[2][8], lst[2][8];
#pragma unroll
    for (int s = 0; s < 2; ++s)
#pragma unroll
        for (int r = 0; r < 8; ++r) { mst[s][r] = -1e30f; lst[s][r] = 0.0f; }

    const float scale = 0.125f;  // 1/sqrt(64)

    const int rK = tid >> 3, cK = tid & 7;  // K tiles: 32 rows x 8 x 16B
    const int rV = tid >> 2, cV = tid & 3;  // V tiles: 64 rows x 4 x 16B

    // 4 async (or sync) 16B transfers per thread per tile
    auto issue_tile = [&](int t) {
        const int buf = t & 1;
        const int j0  = t * A_KT;
        copy16_to_lds(&Ks[buf][0][rK][cK * 8], Kp + (size_t)(j0 + rK) * 64 + cK * 8);
        copy16_to_lds(&Ks[buf][1][rK][cK * 8], Kc + (size_t)(j0 + rK) * 64 + cK * 8);
        copy16_to_lds(&Vts[buf][0][rV][cV * 8], Vp + (size_t)rV * 2048 + j0 + cV * 8);
        copy16_to_lds(&Vts[buf][1][rV][cV * 8], Vc + (size_t)rV * 2048 + j0 + cV * 8);
    };

    issue_tile(0);

    for (int t = 0; t < A_NT; ++t) {
        const bool more = (t + 1) < A_NT;
        if (more) issue_tile(t + 1);       // prefetch next tile into other buffer
        wait_tile_done(more);              // retire this tile's async ops (in-order)
        __syncthreads();                   // all waves' tile data visible in LDS

        const int buf = t & 1;
#pragma unroll
        for (int ai = 0; ai < 2; ++ai) {
            // S = Q K^T for 16 q-rows x 32 keys (two 16x16 C tiles)
            v8f S0 = vzero8(), S1 = vzero8();
#pragma unroll
            for (int dc = 0; dc < 2; ++dc) {
                AF b0, b1;
                b0.h[0] = *reinterpret_cast<const v8h*>(&Ks[buf][ai][lo][dc * 32 + hi * 8]);
                b0.h[1] = *reinterpret_cast<const v8h*>(&Ks[buf][ai][lo][dc * 32 + 16 + hi * 8]);
                b1.h[0] = *reinterpret_cast<const v8h*>(&Ks[buf][ai][16 + lo][dc * 32 + hi * 8]);
                b1.h[1] = *reinterpret_cast<const v8h*>(&Ks[buf][ai][16 + lo][dc * 32 + 16 + hi * 8]);
                S0 = wmma16(qa[ai][dc], b0, S0);
                S1 = wmma16(qa[ai][dc], b1, S1);
            }

            auto consume = [&](v8f (&Ot)[4], const _Float16 (*Vt)[A_KT]) {
                // online softmax update (row = r + 8*hi, cols lo and 16+lo)
#pragma unroll
                for (int r = 0; r < 8; ++r) {
                    float v0 = S0[r] * scale, v1 = S1[r] * scale;
                    float tmax = fmaxf(v0, v1);
#pragma unroll
                    for (int msk = 1; msk < 16; msk <<= 1)
                        tmax = fmaxf(tmax, __shfl_xor(tmax, msk, 32));
                    float mold = mst[ai][r];
                    float mnew = fmaxf(mold, tmax);
                    float alpha = __expf(mold - mnew);
                    float p0 = __expf(v0 - mnew), p1 = __expf(v1 - mnew);
                    float rs = p0 + p1;
#pragma unroll
                    for (int msk = 1; msk < 16; msk <<= 1)
                        rs += __shfl_xor(rs, msk, 32);
                    lst[ai][r] = lst[ai][r] * alpha + rs;
                    mst[ai][r] = mnew;
                    Ot[0][r] *= alpha; Ot[1][r] *= alpha;
                    Ot[2][r] *= alpha; Ot[3][r] *= alpha;
                    Pbuf[wave][r + hi * 8][lo]      = (_Float16)p0;
                    Pbuf[wave][r + hi * 8][16 + lo] = (_Float16)p1;
                }
                // O += P x V  (P via LDS round-trip into A layout)
                AF pa;
                pa.h[0] = *reinterpret_cast<const v8h*>(&Pbuf[wave][lo][hi * 8]);
                pa.h[1] = *reinterpret_cast<const v8h*>(&Pbuf[wave][lo][16 + hi * 8]);
#pragma unroll
                for (int nt = 0; nt < 4; ++nt) {
                    AF bf;
                    bf.h[0] = *reinterpret_cast<const v8h*>(&Vt[nt * 16 + lo][hi * 8]);
                    bf.h[1] = *reinterpret_cast<const v8h*>(&Vt[nt * 16 + lo][16 + hi * 8]);
                    Ot[nt] = wmma16(pa, bf, Ot[nt]);
                }
            };

            if (ai == a1) consume(O1, Vts[buf][0]);   // feeds v1, multiplies Pv
            else          consume(O2, Vts[buf][1]);   // feeds v2, multiplies Cv
        }
        __syncthreads();   // reads done before this buffer is overwritten
    }

    // normalize + store (v1 then v2, f32)
    float* __restrict__ v1o = outp;
    float* __restrict__ v2o = outp + 2ull * 2048 * 1024;
#pragma unroll
    for (int r = 0; r < 8; ++r) {
        int q = q0 + wave * 16 + r + hi * 8;
        float inv1 = 1.0f / lst[a1][r];
        float inv2 = 1.0f / lst[1 - a1][r];
#pragma unroll
        for (int nt = 0; nt < 4; ++nt) {
            int col = h * 64 + nt * 16 + lo;
            size_t idx = ((size_t)b * 2048 + q) * 1024 + col;
            v1o[idx] = O1[nt][r] * inv1;
            v2o[idx] = O2[nt][r] * inv2;
        }
    }
}

extern "C" void kernel_launch(void* const* d_in, const int* in_sizes, int n_in,
                              void* d_out, int out_size, void* d_ws, size_t ws_size,
                              hipStream_t stream) {
    (void)in_sizes; (void)n_in; (void)out_size; (void)ws_size; // ws needs 48 MB
    const float* x = (const float*)d_in[0];
    _Float16* ws = (_Float16*)d_ws;
    float* out = (float*)d_out;

    // proj order: Pq, Pk, Pv, Cq, Ck, Cv ; V-type projections stored transposed
    for (int p = 0; p < 6; ++p) {
        const float* W = (const float*)d_in[1 + p];
        const int isV = (p == 2 || p == 5) ? 1 : 0;
        proj_kernel<<<dim3(8, 32), 256, 0, stream>>>(x, W, ws + (size_t)p * 4096 * 1024, isV);
    }
    attn_kernel<<<dim3(16, 32), 256, 0, stream>>>(ws, out);
}
